// LocalDilatedAttention_18597208392429
// MI455X (gfx1250) — matrix operand order repaired and checked by reading, not compiled, not run on hardware
//
#include <hip/hip_runtime.h>
#include <hip/hip_bf16.h>

typedef _Float16 f16;
typedef _Float16 h4  __attribute__((ext_vector_type(4)));
typedef _Float16 h8  __attribute__((ext_vector_type(8)));
typedef _Float16 v16h __attribute__((ext_vector_type(16)));
typedef float    v8f  __attribute__((ext_vector_type(8)));

#define L_SEQ   4096
#define D_MODEL 512
#define N_HEADS 8
#define D_HEAD  64
#define WINSZ   256
#define NT      (D_MODEL / 16)   // 32 n-tiles
#define KCN     (D_MODEL / 32)   // 16 k-chunks
#define VT_STR  48               // V^T LDS row stride in halves (96B, 32B-aligned rows)

__device__ __forceinline__ v16h make_a(h8 lo, h8 hi) {
  union { v16h v; h8 h[2]; } u;
  u.h[0] = lo; u.h[1] = hi;
  return u.v;
}

// ---------------------------------------------------------------------------
// RMSNorm: x[4096,512] f32 -> qn, kn f16 (one wave per row)
// ---------------------------------------------------------------------------
__global__ __launch_bounds__(32) void rmsnorm_kernel(
    const float* __restrict__ x, const float* __restrict__ gq,
    const float* __restrict__ gkv, f16* __restrict__ qn, f16* __restrict__ kn) {
  int row = blockIdx.x, l = threadIdx.x;
  const float4* xr = (const float4*)(x + (size_t)row * D_MODEL);
  float4 v[4];
  float ss = 0.f;
#pragma unroll
  for (int i = 0; i < 4; ++i) {
    v[i] = xr[l + 32 * i];
    ss += v[i].x * v[i].x + v[i].y * v[i].y + v[i].z * v[i].z + v[i].w * v[i].w;
  }
#pragma unroll
  for (int m = 16; m >= 1; m >>= 1) ss += __shfl_xor(ss, m, 32);
  float r = rsqrtf(ss * (1.f / D_MODEL) + 1e-6f);
#pragma unroll
  for (int i = 0; i < 4; ++i) {
    int base = (l + 32 * i) * 4;
    float4 g4 = ((const float4*)gq)[l + 32 * i];
    float4 k4 = ((const float4*)gkv)[l + 32 * i];
    h4 q = { (f16)(g4.x * v[i].x * r), (f16)(g4.y * v[i].y * r),
             (f16)(g4.z * v[i].z * r), (f16)(g4.w * v[i].w * r) };
    h4 k = { (f16)(k4.x * v[i].x * r), (f16)(k4.y * v[i].y * r),
             (f16)(k4.z * v[i].z * r), (f16)(k4.w * v[i].w * r) };
    *(h4*)(qn + (size_t)row * D_MODEL + base) = q;
    *(h4*)(kn + (size_t)row * D_MODEL + base) = k;
  }
}

// ---------------------------------------------------------------------------
// Pack W[512,512] f32 into WMMA B-operand layout, f16.
// B element (K=k, N=n) = W[n][k]; lane l holds N=l&15, K=(l>>4)*16+e.
// ---------------------------------------------------------------------------
__global__ __launch_bounds__(32) void pack_w_kernel(
    const float* __restrict__ W, f16* __restrict__ Wp) {
  int kc = blockIdx.x, nt = blockIdx.y, l = threadIdx.x;
  int n  = nt * 16 + (l & 15);
  int kb = kc * 32 + (l >> 4) * 16;
  const float* src = W + (size_t)n * D_MODEL + kb;
  v16h out;
#pragma unroll
  for (int e = 0; e < 16; ++e) out[e] = (f16)src[e];
  *(v16h*)(Wp + (size_t)((kc * NT + nt) * 32 + l) * 16) = out;
}

// ---------------------------------------------------------------------------
// GEMM: C[m,n] = sum_k A[m,k] * W[n,k];  A f16 row-major [M,512], Wp packed.
// One wave per 16x64 output block (4 n-tiles): A loaded once per k-chunk,
// 4 WMMAs share it -> 64 v_wmma_f32_16x16x32_f16 per wave, 4x less A traffic.
// ---------------------------------------------------------------------------
template <bool F32OUT>
__global__ __launch_bounds__(32) void gemm_kernel(
    const f16* __restrict__ A, const f16* __restrict__ Wp,
    f16* __restrict__ outh, float* __restrict__ outf) {
  int mt = blockIdx.x, ng = blockIdx.y, l = threadIdx.x;
  int lh = l & 15, hf = l >> 4;
  const f16* arow  = A + (size_t)(mt * 16 + lh) * D_MODEL;
  const f16* wbase = Wp + (size_t)(ng * 4) * 32 * 16 + (size_t)l * 16;
  v8f c0 = {}, c1 = {}, c2 = {}, c3 = {};
#pragma unroll
  for (int kc = 0; kc < KCN; ++kc) {
    h8 lo = *(const h8*)(arow + kc * 32 + hf * 16);
    h8 hi = *(const h8*)(arow + kc * 32 + hf * 16 + 8);
    v16h a = make_a(lo, hi);
    const f16* wk = wbase + (size_t)kc * (NT * 32 * 16);
    v16h b0 = *(const v16h*)(wk + 0 * 512);
    v16h b1 = *(const v16h*)(wk + 1 * 512);
    v16h b2 = *(const v16h*)(wk + 2 * 512);
    v16h b3 = *(const v16h*)(wk + 3 * 512);
    c0 = __builtin_amdgcn_wmma_f32_16x16x32_f16(false, a, false, b0, (short)0, c0, false, false);
    c1 = __builtin_amdgcn_wmma_f32_16x16x32_f16(false, a, false, b1, (short)0, c1, false, false);
    c2 = __builtin_amdgcn_wmma_f32_16x16x32_f16(false, a, false, b2, (short)0, c2, false, false);
    c3 = __builtin_amdgcn_wmma_f32_16x16x32_f16(false, a, false, b3, (short)0, c3, false, false);
  }
#pragma unroll
  for (int v = 0; v < 8; ++v) {
    size_t m = (size_t)(mt * 16 + v + 8 * hf);
    int n0 = ng * 64 + lh;
    if (F32OUT) {
      outf[m * D_MODEL + n0 +  0] = c0[v];
      outf[m * D_MODEL + n0 + 16] = c1[v];
      outf[m * D_MODEL + n0 + 32] = c2[v];
      outf[m * D_MODEL + n0 + 48] = c3[v];
    } else {
      outh[m * D_MODEL + n0 +  0] = (f16)c0[v];
      outh[m * D_MODEL + n0 + 16] = (f16)c1[v];
      outh[m * D_MODEL + n0 + 32] = (f16)c2[v];
      outh[m * D_MODEL + n0 + 48] = (f16)c3[v];
    }
  }
}

// ---------------------------------------------------------------------------
// Banded flash attention (one wave per 16-row q-tile per head).
// dilated=0: |j-i|<=256 ; dilated=1: additionally (j-i)%4==0.
// K tile staged via GLOBAL_LOAD_ASYNC_TO_LDS_B128 (ASYNCcnt path).
// ---------------------------------------------------------------------------
__global__ __launch_bounds__(32) void attn_kernel(
    const f16* __restrict__ Qf, const f16* __restrict__ Kf,
    const f16* __restrict__ Vf, const unsigned char* __restrict__ kpm,
    f16* __restrict__ O, int dilated) {
  __shared__ __align__(32) f16 lk[32 * 64];        // K tile, row-major
  __shared__ __align__(32) f16 lvt[64 * VT_STR];   // V^T tile [d][j]
  __shared__ __align__(32) f16 lp[16 * 32];        // P tile (C->A relayout)

  const float NEG_INF = -__builtin_inff();
  int h = blockIdx.y, qt = blockIdx.x, l = threadIdx.x;
  int lh = l & 15, hf = l >> 4;
  int m0 = qt * 16;

  // Q tile in A-operand layout, kept resident (two 32-wide k-chunks of DH=64)
  const f16* qrow = Qf + (size_t)(m0 + lh) * D_MODEL + h * D_HEAD;
  v16h aQ0 = make_a(*(const h8*)(qrow + hf * 16),
                    *(const h8*)(qrow + hf * 16 + 8));
  v16h aQ1 = make_a(*(const h8*)(qrow + 32 + hf * 16),
                    *(const h8*)(qrow + 32 + hf * 16 + 8));

  v8f o0 = {}, o1 = {}, o2 = {}, o3 = {};
  float mrow[8], lrow[8];
#pragma unroll
  for (int v = 0; v < 8; ++v) { mrow[v] = NEG_INF; lrow[v] = 0.f; }

  int jlo = m0 - WINSZ; if (jlo < 0) jlo = 0; jlo &= ~31;
  int jhi = m0 + 15 + WINSZ; if (jhi > L_SEQ - 1) jhi = L_SEQ - 1;

  unsigned lk_base = (unsigned)(size_t)(lk + l * 64);

  for (int jb = jlo; jb <= jhi; jb += 32) {
    int j = jb + l;  // always < L (jhi <= 4095, 32-aligned up stays in range)
    const f16* krow = Kf + (size_t)j * D_MODEL + h * D_HEAD;
    const f16* vrow = Vf + (size_t)j * D_MODEL + h * D_HEAD;

    // Make sure prior-iteration LDS reads retired before the DMA overwrites lk,
    // then async-copy this lane's K row (128B) straight into LDS.
    asm volatile("s_wait_dscnt 0x0" ::: "memory");
#pragma unroll
    for (int i = 0; i < 8; ++i) {
      unsigned long long ga = (unsigned long long)(size_t)(krow + i * 8);
      asm volatile("global_load_async_to_lds_b128 %0, %1, off"
                   :: "v"(lk_base + i * 16), "v"(ga) : "memory");
    }
    // V row -> registers -> transposed scatter into lvt
#pragma unroll
    for (int i = 0; i < 8; ++i) {
      h8 vv = *(const h8*)(vrow + i * 8);
#pragma unroll
      for (int e = 0; e < 8; ++e) lvt[(i * 8 + e) * VT_STR + l] = vv[e];
    }
    if (jb + 32 <= jhi) {  // prefetch next K/V rows -> global_prefetch_b8
      __builtin_prefetch(Kf + (size_t)(jb + 32 + l) * D_MODEL + h * D_HEAD, 0, 3);
      __builtin_prefetch(Vf + (size_t)(jb + 32 + l) * D_MODEL + h * D_HEAD, 0, 3);
    }
    asm volatile("s_wait_asynccnt 0x0" ::: "memory");
    __syncthreads();

    // S = (Q K^T) * scale over two 16-key tiles, masked
    float sv[2][8];
#pragma unroll
    for (int t = 0; t < 2; ++t) {
      v16h b0 = *(const v16h*)(lk + (t * 16 + lh) * 64 + hf * 16);
      v16h b1 = *(const v16h*)(lk + (t * 16 + lh) * 64 + 32 + hf * 16);
      v8f s = {};
      s = __builtin_amdgcn_wmma_f32_16x16x32_f16(false, aQ0, false, b0,
                                                 (short)0, s, false, false);
      s = __builtin_amdgcn_wmma_f32_16x16x32_f16(false, aQ1, false, b1,
                                                 (short)0, s, false, false);
      int jj = jb + t * 16 + lh;
      bool kp = (kpm[jj] == 0);
#pragma unroll
      for (int v = 0; v < 8; ++v) {
        int m = m0 + v + 8 * hf;
        int rel = jj - m;
        bool keep = kp && (rel >= -WINSZ) && (rel <= WINSZ) &&
                    (!dilated || ((rel & 3) == 0));
        sv[t][v] = keep ? (s[v] * 0.125f) : NEG_INF;
      }
    }

    // online softmax (row lives across 16 lanes of one half-wave)
    float alpha[8];
#pragma unroll
    for (int v = 0; v < 8; ++v) {
      float rm = fmaxf(sv[0][v], sv[1][v]);
#pragma unroll
      for (int msk = 8; msk >= 1; msk >>= 1) rm = fmaxf(rm, __shfl_xor(rm, msk, 32));
      float mn  = fmaxf(mrow[v], rm);
      float mns = (mn < -1e37f) ? 0.f : mn;
      alpha[v] = __expf(mrow[v] - mns);  // exp(-inf) = 0 on first touch
      mrow[v]  = mn;
      float p0 = __expf(sv[0][v] - mns);
      float p1 = __expf(sv[1][v] - mns);
      lp[(v + 8 * hf) * 32 + lh]      = (f16)p0;
      lp[(v + 8 * hf) * 32 + 16 + lh] = (f16)p1;
      float rs = p0 + p1;
#pragma unroll
      for (int msk = 8; msk >= 1; msk >>= 1) rs += __shfl_xor(rs, msk, 32);
      lrow[v] = lrow[v] * alpha[v] + rs;
    }
#pragma unroll
    for (int v = 0; v < 8; ++v) {
      o0[v] *= alpha[v]; o1[v] *= alpha[v]; o2[v] *= alpha[v]; o3[v] *= alpha[v];
    }
    __syncthreads();

    // O += P * V  (P relayed through LDS into A-operand layout)
    v16h aP = make_a(*(const h8*)(lp + lh * 32 + hf * 16),
                     *(const h8*)(lp + lh * 32 + hf * 16 + 8));
    v16h bv0 = *(const v16h*)(lvt + ( 0 + lh) * VT_STR + hf * 16);
    v16h bv1 = *(const v16h*)(lvt + (16 + lh) * VT_STR + hf * 16);
    v16h bv2 = *(const v16h*)(lvt + (32 + lh) * VT_STR + hf * 16);
    v16h bv3 = *(const v16h*)(lvt + (48 + lh) * VT_STR + hf * 16);
    o0 = __builtin_amdgcn_wmma_f32_16x16x32_f16(false, aP, false, bv0, (short)0, o0, false, false);
    o1 = __builtin_amdgcn_wmma_f32_16x16x32_f16(false, aP, false, bv1, (short)0, o1, false, false);
    o2 = __builtin_amdgcn_wmma_f32_16x16x32_f16(false, aP, false, bv2, (short)0, o2, false, false);
    o3 = __builtin_amdgcn_wmma_f32_16x16x32_f16(false, aP, false, bv3, (short)0, o3, false, false);
    __syncthreads();
  }

#pragma unroll
  for (int v = 0; v < 8; ++v) lrow[v] = 1.f / lrow[v];
#pragma unroll
  for (int v = 0; v < 8; ++v) {
    size_t rbase = (size_t)(m0 + v + 8 * hf) * D_MODEL + h * D_HEAD + lh;
    O[rbase +  0] = (f16)(o0[v] * lrow[v]);
    O[rbase + 16] = (f16)(o1[v] * lrow[v]);
    O[rbase + 32] = (f16)(o2[v] * lrow[v]);
    O[rbase + 48] = (f16)(o3[v] * lrow[v]);
  }
}

// ---------------------------------------------------------------------------
extern "C" void kernel_launch(void* const* d_in, const int* in_sizes, int n_in,
                              void* d_out, int out_size, void* d_ws, size_t ws_size,
                              hipStream_t stream) {
  (void)in_sizes; (void)n_in; (void)out_size; (void)ws_size;
  const float* x   = (const float*)d_in[0];
  const unsigned char* kpm = (const unsigned char*)d_in[1];
  const float* W[8];
  for (int i = 0; i < 8; ++i) W[i] = (const float*)d_in[2 + i];
  const float* gq  = (const float*)d_in[10];
  const float* gkv = (const float*)d_in[11];
  float* outf = (float*)d_out;

  char* ws = (char*)d_ws;
  const size_t MB = 1ull << 20;
  f16* qn = (f16*)(ws + 0 * MB);
  f16* kn = (f16*)(ws + 4 * MB);
  f16* wp[8];
  for (int i = 0; i < 8; ++i) wp[i] = (f16*)(ws + 8 * MB + (size_t)i * 512 * 1024);
  f16* proj[6];  // Ql,Kl,Vl,Qd,Kd,Vd
  for (int i = 0; i < 6; ++i) proj[i] = (f16*)(ws + 12 * MB + (size_t)i * 4 * MB);
  f16* attn[2];
  for (int i = 0; i < 2; ++i) attn[i] = (f16*)(ws + 36 * MB + (size_t)i * 4 * MB);

  rmsnorm_kernel<<<L_SEQ, 32, 0, stream>>>(x, gq, gkv, qn, kn);
  for (int i = 0; i < 8; ++i)
    pack_w_kernel<<<dim3(KCN, NT), 32, 0, stream>>>(W[i], wp[i]);

  dim3 gg(L_SEQ / 16, NT / 4);   // 4 n-tiles per wave
  gemm_kernel<false><<<gg, 32, 0, stream>>>(qn, wp[0], proj[0], nullptr); // Q_l
  gemm_kernel<false><<<gg, 32, 0, stream>>>(kn, wp[1], proj[1], nullptr); // K_l
  gemm_kernel<false><<<gg, 32, 0, stream>>>(kn, wp[2], proj[2], nullptr); // V_l
  gemm_kernel<false><<<gg, 32, 0, stream>>>(qn, wp[4], proj[3], nullptr); // Q_d
  gemm_kernel<false><<<gg, 32, 0, stream>>>(kn, wp[5], proj[4], nullptr); // K_d
  gemm_kernel<false><<<gg, 32, 0, stream>>>(kn, wp[6], proj[5], nullptr); // V_d

  dim3 ga(L_SEQ / 16, N_HEADS);
  attn_kernel<<<ga, 32, 0, stream>>>(proj[0], proj[1], proj[2], kpm, attn[0], 0);
  attn_kernel<<<ga, 32, 0, stream>>>(proj[3], proj[4], proj[5], kpm, attn[1], 1);

  gemm_kernel<true><<<gg, 32, 0, stream>>>(attn[0], wp[3], nullptr, outf);
  gemm_kernel<true><<<gg, 32, 0, stream>>>(attn[1], wp[7], nullptr,
                                           outf + (size_t)L_SEQ * D_MODEL);
}
